// XCAttention_16114717295153
// MI455X (gfx1250) — matrix operand order repaired and checked by reading, not compiled, not run on hardware
//
#include <hip/hip_runtime.h>
#include <hip/hip_bf16.h>
#include <math.h>

// XCA attention, MI455X (gfx1250, wave32, WMMA + TDM).
//   B=8, N=4096, C=768, H=32, D=24
// GEMMs: v_wmma_f32_16x16x32_bf16, operands pre-converted to bf16 in global,
// tiles staged LDS via tensor_load_to_lds (TENSORcnt), double-buffered.
//
// Workspace (bytes):
//   [0)                qkvT f32: 3 slabs (q,k,v), each SLAB=25165824 floats
//                      (slab 0 reused after K2 as ys_bf16, 50 MB)
//   [+301989888)       attn f32 (B*H*D*D = 147456 floats)
//   [+302579712)       x_bf16   (25165824 bf16)
//   [+352911360)       WqkvT_bf16 (2304x768)
//   [+356450304)       WpT_bf16   (768x768)
//   total ~357.6 MB

#define BB 8
#define NN 4096
#define CC 768
#define HH 32
#define DD 24
#define SLAB (BB * CC * NN)

typedef __attribute__((ext_vector_type(16))) __bf16 v16bf;
typedef __attribute__((ext_vector_type(4)))  __bf16 v4bf;
typedef __attribute__((ext_vector_type(8)))  float  v8f;
typedef __attribute__((ext_vector_type(4)))  unsigned int v4u;
typedef __attribute__((ext_vector_type(8)))  int v8i;
typedef __attribute__((ext_vector_type(4)))  int v4i;

union Frag16 { v4u u[2]; v16bf v; };   // 16 bf16 = 2 dwordx4

#if __has_builtin(__builtin_amdgcn_tensor_load_to_lds)
#define USE_TDM 1
#else
#define USE_TDM 0
#endif

// ---------------------------------------------------------------------------
// TDM: load a (rows x 32) bf16 tile, row stride `stride_elems`, into LDS with
// 16B pad after each 64B row (-> 80B LDS row stride, matches [..][40] arrays).
// D# per CDNA5 ISA 8.3/8.4: group0 {count=1, lds_addr, global_addr, type=2},
// group1 {data_size=2B, pad_enable, pad_interval=16dw, pad_amount=4dw, dims}.
// This toolchain's builtin is the 6-arg form:
//   (uint32x4 g0, int32x8 g1, int32x4 g2, int32x4 g3, int32x8, i32 cpol)
// ---------------------------------------------------------------------------
__device__ __forceinline__ void tdm_load_tile_bf16(
    unsigned lds_byte_addr, const __bf16* g, int rows,
    unsigned long long stride_elems) {
#if USE_TDM
  const unsigned long long ga = (unsigned long long)(uintptr_t)g;
  v4u g0;
  g0[0] = 1u;                                            // count=1, user mode
  g0[1] = lds_byte_addr;                                 // lds_addr
  g0[2] = (unsigned)ga;                                  // global_addr[31:0]
  g0[3] = (unsigned)((ga >> 32) & 0x01FFFFFFull) | (2u << 30);  // [56:32]|type=2
  v8i g1;
  g1[0] = (int)((1u << 16) | (1u << 20) | (3u << 22) | (3u << 25));
  const unsigned tw = 1u << 20, th = 1u << 20;           // huge dims: no OOB clip
  g1[1] = (int)((tw & 0xFFFFu) << 16);
  g1[2] = (int)((tw >> 16) | ((th & 0xFFFFu) << 16));
  g1[3] = (int)((th >> 16) | (32u << 16));               // tile_dim0 = 32
  g1[4] = rows;                                          // tile_dim1 (tile_dim2=0)
  g1[5] = (int)(unsigned)stride_elems;
  g1[6] = (int)(unsigned)(stride_elems >> 32);
  g1[7] = 0;
  const v4i z4 = {0, 0, 0, 0};
  const v8i z8 = {0, 0, 0, 0, 0, 0, 0, 0};
  __builtin_amdgcn_tensor_load_to_lds(g0, g1, z4, z4, z8, 0);
#endif
}

__device__ __forceinline__ void fallback_stage(
    __bf16* lds, const __bf16* g, int rows, int stride, int t) {
#if !USE_TDM
  // per-lane staging: rows x 32 bf16, LDS rows padded to 40
  for (int i = t; i < rows * 16; i += 256) {
    const int r = i >> 4, u = i & 15;
    *(unsigned*)&lds[r * 40 + u * 2] = *(const unsigned*)&g[r * stride + u * 2];
  }
#endif
}

// ---------------------------------------------------------------------------
// K0: elementwise f32 -> bf16
// ---------------------------------------------------------------------------
__global__ __launch_bounds__(256) void cvt_kernel(
    const float* __restrict__ in, __bf16* __restrict__ out, int n4) {
  const int i = blockIdx.x * 256 + threadIdx.x;
  if (i < n4) {
    const float4 v = ((const float4*)in)[i];
    v4bf o;
    o[0] = (__bf16)v.x; o[1] = (__bf16)v.y; o[2] = (__bf16)v.z; o[3] = (__bf16)v.w;
    ((v4bf*)out)[i] = o;
  }
}

// K0: transpose + cvt: W (R x Cc) -> WT (Cc x R), WT[c][r] = W[r][c]
__global__ __launch_bounds__(256) void transpose_cvt_kernel(
    const float* __restrict__ W, __bf16* __restrict__ WT, int R, int Cc) {
  const int c = blockIdx.x * 256 + threadIdx.x;
  const int r = blockIdx.y;
  if (c < Cc) WT[(size_t)c * R + r] = (__bf16)W[(size_t)r * Cc + c];
}

// ---------------------------------------------------------------------------
// K1: qkv^T GEMM: qkvT[j][m] = sum_c WqkvT_bf[j][c]*x_bf[m][c] + b[j]
// M=j(2304), N=m(32768), K=768.  Block tile 128j x 256m, 8 waves,
// wave = 32j x 128m = 2 A-frags x 8 B-frags = 16 WMMA / K-step.
// Stores land directly in (B,H,D,N) q/k/v slabs (tokens contiguous).
// ---------------------------------------------------------------------------
__global__ __launch_bounds__(256) void qkv_gemm_kernel(
    const __bf16* __restrict__ xbf, const __bf16* __restrict__ WqkvT,
    const float* __restrict__ bqkv, float* __restrict__ qkvT) {
  __shared__ __align__(16) __bf16 Alds[2][128][40];
  __shared__ __align__(16) __bf16 Blds[2][256][40];
  const int t = threadIdx.x;
  const int jTile = blockIdx.x * 128;
  const int mTile = blockIdx.y * 256;
  const int lane = t & 31;
  const int w = t >> 5;
  const int wjb = w & 3;                 // 32-wide j block
  const int wmb = w >> 2;                // 128-wide m half
  const int nloc = lane & 15;
  const int ka = (lane < 16) ? 0 : 8;    // A frag K {0-7,16-23}/{8-15,24-31}
  const int kb = (lane < 16) ? 0 : 8;    // B frag K {0-7,16-23}/{8-15,24-31}
  const __bf16* gA = WqkvT + (size_t)jTile * CC;
  const __bf16* gB = xbf + (size_t)mTile * CC;

  v8f acc[2][8];
#pragma unroll
  for (int ia = 0; ia < 2; ++ia)
#pragma unroll
    for (int mb = 0; mb < 8; ++mb) acc[ia][mb] = v8f{0.f};

  const int NIT = CC / 32;
#if USE_TDM
  if (w == 0) {
    tdm_load_tile_bf16((unsigned)(uintptr_t)&Alds[0][0][0], gA, 128, CC);
    tdm_load_tile_bf16((unsigned)(uintptr_t)&Blds[0][0][0], gB, 256, CC);
    __builtin_amdgcn_s_wait_tensorcnt(0);
  }
#else
  fallback_stage(&Alds[0][0][0], gA, 128, CC, t);
  fallback_stage(&Blds[0][0][0], gB, 256, CC, t);
#endif
  __syncthreads();

  for (int it = 0; it < NIT; ++it) {
    const int cur = it & 1;
    if (it + 1 < NIT) {
#if USE_TDM
      if (w == 0) {
        tdm_load_tile_bf16((unsigned)(uintptr_t)&Alds[1 - cur][0][0],
                           gA + (it + 1) * 32, 128, CC);
        tdm_load_tile_bf16((unsigned)(uintptr_t)&Blds[1 - cur][0][0],
                           gB + (it + 1) * 32, 256, CC);
      }
#else
      fallback_stage(&Alds[1 - cur][0][0], gA + (it + 1) * 32, 128, CC, t);
      fallback_stage(&Blds[1 - cur][0][0], gB + (it + 1) * 32, 256, CC, t);
#endif
    }
    Frag16 a0, a1;
    const int ar = wjb * 32 + nloc;
    a0.u[0] = *(const v4u*)&Alds[cur][ar][ka];
    a0.u[1] = *(const v4u*)&Alds[cur][ar][ka + 16];
    a1.u[0] = *(const v4u*)&Alds[cur][ar + 16][ka];
    a1.u[1] = *(const v4u*)&Alds[cur][ar + 16][ka + 16];
#pragma unroll
    for (int mb = 0; mb < 8; ++mb) {
      Frag16 bf;
      const int br = wmb * 128 + mb * 16 + nloc;
      bf.u[0] = *(const v4u*)&Blds[cur][br][kb];
      bf.u[1] = *(const v4u*)&Blds[cur][br][kb + 16];
      acc[0][mb] = __builtin_amdgcn_wmma_f32_16x16x32_bf16(
          false, a0.v, false, bf.v, (short)0, acc[0][mb], false, false);
      acc[1][mb] = __builtin_amdgcn_wmma_f32_16x16x32_bf16(
          false, a1.v, false, bf.v, (short)0, acc[1][mb], false, false);
    }
#if USE_TDM
    if (w == 0 && it + 1 < NIT) __builtin_amdgcn_s_wait_tensorcnt(0);
#endif
    __syncthreads();
  }

  // Store. jTile (mult of 128) never crosses a q/k/v boundary (768%128==0).
  const int three = jTile / CC;
  const int jjB = jTile - three * CC;
  const int hs = (lane < 16) ? 0 : 8;
  const int bidx = mTile >> 12;          // token batch; tile never crosses b
  const int n0 = (mTile & (NN - 1)) + wmb * 128 + nloc;
  float* outS = qkvT + (size_t)three * SLAB + (size_t)bidx * CC * NN;
  float bias[2][8];
#pragma unroll
  for (int ia = 0; ia < 2; ++ia)
#pragma unroll
    for (int i = 0; i < 8; ++i)
      bias[ia][i] = bqkv[jTile + wjb * 32 + ia * 16 + i + hs];
#pragma unroll
  for (int ia = 0; ia < 2; ++ia) {
#pragma unroll
    for (int i = 0; i < 8; ++i) {
      const int jj = jjB + wjb * 32 + ia * 16 + i + hs;
      float* row = outS + (size_t)jj * NN + n0;
#pragma unroll
      for (int mb = 0; mb < 8; ++mb)
        row[mb * 16] = acc[ia][mb][i] + bias[ia][i];
    }
  }
}

// ---------------------------------------------------------------------------
// K2: per (b,h): sumsq(q_d), sumsq(k_e), gram[d][e] over N, then
// attn = softmax(gram/(max(|q|,eps)*max(|k|,eps)) * temperature[h]).  fp32 VALU.
// ---------------------------------------------------------------------------
__global__ __launch_bounds__(256) void attn_kernel(
    const float* __restrict__ qkvT, const float* __restrict__ temperature,
    float* __restrict__ attn) {
  __shared__ float qs[DD][129];
  __shared__ float ks[DD][129];
  __shared__ float gsm[576 + 48];
  const int t = threadIdx.x;
  const int bh = blockIdx.x;
  const int b = bh >> 5, h = bh & 31;
  const float* qb = qkvT + (size_t)(b * CC + h * DD) * NN;
  const float* kb = qkvT + (size_t)SLAB + (size_t)(b * CC + h * DD) * NN;

  const int p0 = t, p1 = t + 256, p2 = t + 512;
  const int d0 = p0 / DD, e0 = p0 % DD;
  const int d1 = p1 / DD, e1 = p1 % DD;
  const bool has2 = (p2 < 576);
  const int d2 = has2 ? p2 / DD : 0, e2 = has2 ? p2 % DD : 0;
  float a0 = 0.f, a1 = 0.f, a2 = 0.f, asq = 0.f;

  for (int c = 0; c < NN; c += 128) {
    for (int i = t; i < DD * 128; i += 256) {
      const int d = i >> 7, nl = i & 127;
      qs[d][nl] = qb[(size_t)d * NN + c + nl];
      ks[d][nl] = kb[(size_t)d * NN + c + nl];
    }
    __syncthreads();
    for (int nl = 0; nl < 128; ++nl) {
      a0 = fmaf(qs[d0][nl], ks[e0][nl], a0);
      a1 = fmaf(qs[d1][nl], ks[e1][nl], a1);
      if (has2) a2 = fmaf(qs[d2][nl], ks[e2][nl], a2);
      if (t < DD) { const float qv = qs[t][nl]; asq = fmaf(qv, qv, asq); }
      else if (t >= 32 && t < 32 + DD) { const float kv = ks[t - 32][nl]; asq = fmaf(kv, kv, asq); }
    }
    __syncthreads();
  }
  gsm[p0] = a0;
  gsm[p1] = a1;
  if (has2) gsm[p2] = a2;
  if (t < DD) gsm[576 + t] = asq;
  else if (t >= 32 && t < 32 + DD) gsm[600 + (t - 32)] = asq;
  __syncthreads();

  if (t < DD) {
    const float nq = fmaxf(sqrtf(gsm[576 + t]), 1e-12f);
    const float tp = temperature[h];
    float l[DD];
    float mx = -3.0e38f;
    for (int e = 0; e < DD; ++e) {
      const float nk = fmaxf(sqrtf(gsm[600 + e]), 1e-12f);
      const float v = gsm[t * DD + e] * tp / (nq * nk);
      l[e] = v;
      mx = fmaxf(mx, v);
    }
    float s = 0.f;
    for (int e = 0; e < DD; ++e) { l[e] = __expf(l[e] - mx); s += l[e]; }
    const float inv = 1.f / s;
    for (int e = 0; e < DD; ++e) attn[(size_t)bh * 576 + t * DD + e] = l[e] * inv;
  }
}

// ---------------------------------------------------------------------------
// K3: y[b,h,d,n] = sum_e attn*v, stored as bf16 at the torch-faithful flat
// position ((h*B+b)*D+d)*N+n == the (B,N,C) matrix K4 consumes as A.
// ---------------------------------------------------------------------------
__global__ __launch_bounds__(256) void av_kernel(
    const float* __restrict__ qkvT, const float* __restrict__ attn,
    __bf16* __restrict__ ysbf) {
  __shared__ float As[DD * DD];
  const int t = threadIdx.x;
  const int bh = blockIdx.x >> 4;
  const int chunk = blockIdx.x & 15;
  const int b = bh >> 5, h = bh & 31;
  for (int i = t; i < DD * DD; i += 256) As[i] = attn[(size_t)bh * 576 + i];
  __syncthreads();
  const int n = chunk * 256 + t;
  const float* vb = qkvT + 2 * (size_t)SLAB + (size_t)(b * CC + h * DD) * NN + n;
  float ve[DD];
#pragma unroll
  for (int e = 0; e < DD; ++e) ve[e] = vb[(size_t)e * NN];
  __bf16* yb = ysbf + (size_t)((h * BB + b) * DD) * NN + n;
#pragma unroll
  for (int d = 0; d < DD; ++d) {
    float a = 0.f;
#pragma unroll
    for (int e = 0; e < DD; ++e) a = fmaf(As[d * DD + e], ve[e], a);
    yb[(size_t)d * NN] = (__bf16)a;
  }
}

// ---------------------------------------------------------------------------
// K4: out[m][j] = sum_k ysbf[m][k]*WpT[j][k] + bp[j]
// M=32768, N=768, K=768. Block 256m x 128j, 8 waves, wave = 32m x 128j.
// ---------------------------------------------------------------------------
__global__ __launch_bounds__(256) void proj_gemm_kernel(
    const __bf16* __restrict__ ysbf, const __bf16* __restrict__ WpT,
    const float* __restrict__ bp, float* __restrict__ out) {
  __shared__ __align__(16) __bf16 Alds[2][256][40];
  __shared__ __align__(16) __bf16 Blds[2][128][40];
  const int t = threadIdx.x;
  const int jTile = blockIdx.x * 128;
  const int mTile = blockIdx.y * 256;
  const int lane = t & 31;
  const int w = t >> 5;
  const int nloc = lane & 15;
  const int ka = (lane < 16) ? 0 : 8;
  const int kb = (lane < 16) ? 0 : 8;
  const __bf16* gA = ysbf + (size_t)mTile * CC;
  const __bf16* gB = WpT + (size_t)jTile * CC;

  v8f acc[2][8];
#pragma unroll
  for (int ia = 0; ia < 2; ++ia)
#pragma unroll
    for (int jb = 0; jb < 8; ++jb) acc[ia][jb] = v8f{0.f};

  const int NIT = CC / 32;
#if USE_TDM
  if (w == 0) {
    tdm_load_tile_bf16((unsigned)(uintptr_t)&Alds[0][0][0], gA, 256, CC);
    tdm_load_tile_bf16((unsigned)(uintptr_t)&Blds[0][0][0], gB, 128, CC);
    __builtin_amdgcn_s_wait_tensorcnt(0);
  }
#else
  fallback_stage(&Alds[0][0][0], gA, 256, CC, t);
  fallback_stage(&Blds[0][0][0], gB, 128, CC, t);
#endif
  __syncthreads();

  for (int it = 0; it < NIT; ++it) {
    const int cur = it & 1;
    if (it + 1 < NIT) {
#if USE_TDM
      if (w == 0) {
        tdm_load_tile_bf16((unsigned)(uintptr_t)&Alds[1 - cur][0][0],
                           gA + (it + 1) * 32, 256, CC);
        tdm_load_tile_bf16((unsigned)(uintptr_t)&Blds[1 - cur][0][0],
                           gB + (it + 1) * 32, 128, CC);
      }
#else
      fallback_stage(&Alds[1 - cur][0][0], gA + (it + 1) * 32, 256, CC, t);
      fallback_stage(&Blds[1 - cur][0][0], gB + (it + 1) * 32, 128, CC, t);
#endif
    }
    Frag16 a0, a1;
    const int ar = w * 32 + nloc;
    a0.u[0] = *(const v4u*)&Alds[cur][ar][ka];
    a0.u[1] = *(const v4u*)&Alds[cur][ar][ka + 16];
    a1.u[0] = *(const v4u*)&Alds[cur][ar + 16][ka];
    a1.u[1] = *(const v4u*)&Alds[cur][ar + 16][ka + 16];
#pragma unroll
    for (int jb = 0; jb < 8; ++jb) {
      Frag16 bf;
      const int br = jb * 16 + nloc;
      bf.u[0] = *(const v4u*)&Blds[cur][br][kb];
      bf.u[1] = *(const v4u*)&Blds[cur][br][kb + 16];
      acc[0][jb] = __builtin_amdgcn_wmma_f32_16x16x32_bf16(
          false, a0.v, false, bf.v, (short)0, acc[0][jb], false, false);
      acc[1][jb] = __builtin_amdgcn_wmma_f32_16x16x32_bf16(
          false, a1.v, false, bf.v, (short)0, acc[1][jb], false, false);
    }
#if USE_TDM
    if (w == 0 && it + 1 < NIT) __builtin_amdgcn_s_wait_tensorcnt(0);
#endif
    __syncthreads();
  }

  const int hs = (lane < 16) ? 0 : 8;
  float bias[8];
#pragma unroll
  for (int jb = 0; jb < 8; ++jb) bias[jb] = bp[jTile + jb * 16 + nloc];
#pragma unroll
  for (int ia = 0; ia < 2; ++ia) {
#pragma unroll
    for (int i = 0; i < 8; ++i) {
      const int m = mTile + w * 32 + ia * 16 + i + hs;
      float* row = out + (size_t)m * CC + jTile + nloc;
#pragma unroll
      for (int jb = 0; jb < 8; ++jb)
        row[jb * 16] = acc[ia][jb][i] + bias[jb];
    }
  }
}

extern "C" void kernel_launch(void* const* d_in, const int* in_sizes, int n_in,
                              void* d_out, int out_size, void* d_ws, size_t ws_size,
                              hipStream_t stream) {
  (void)in_sizes; (void)n_in; (void)out_size; (void)ws_size;
  const float* x    = (const float*)d_in[0];
  const float* Wqkv = (const float*)d_in[1];
  const float* bqkv = (const float*)d_in[2];
  const float* temp = (const float*)d_in[3];
  const float* Wp   = (const float*)d_in[4];
  const float* bp   = (const float*)d_in[5];
  float* out = (float*)d_out;
  char* ws = (char*)d_ws;

  float*  qkvT   = (float*)ws;                          // 3 slabs f32
  float*  attn   = (float*)(ws + 301989888);
  __bf16* xbf    = (__bf16*)(ws + 302579712);
  __bf16* WqkvTb = (__bf16*)(ws + 352911360);
  __bf16* WpTb   = (__bf16*)(ws + 356450304);
  __bf16* ysbf   = (__bf16*)qkvT;                       // reuse Q slab after K2

  // K0: precision conversion / weight transposition (bf16 operand streams)
  cvt_kernel<<<dim3((BB * NN * CC / 4 + 255) / 256), 256, 0, stream>>>(
      x, xbf, BB * NN * CC / 4);
  transpose_cvt_kernel<<<dim3(9, CC), 256, 0, stream>>>(Wqkv, WqkvTb, CC, 3 * CC);
  transpose_cvt_kernel<<<dim3(3, CC), 256, 0, stream>>>(Wp, WpTb, CC, CC);

  // K1..K4
  qkv_gemm_kernel<<<dim3(18, 128), 256, 0, stream>>>(xbf, WqkvTb, bqkv, qkvT);
  attn_kernel<<<dim3(BB * HH), 256, 0, stream>>>(qkvT, temp, attn);
  av_kernel<<<dim3(BB * HH * 16), 256, 0, stream>>>(qkvT, attn, ysbf);
  proj_gemm_kernel<<<dim3(6, 128), 256, 0, stream>>>(ysbf, WpTb, bp, out);
}
